// durationLMLayer_79559974191174
// MI455X (gfx1250) — matrix operand order repaired and checked by reading, not compiled, not run on hardware
//
#include <hip/hip_runtime.h>
#include <hip/hip_bf16.h>
#include <math.h>

// ---------------------------------------------------------------------------
// durationLM on MI455X (gfx1250, wave32, WMMA).
//   B=8, S=256, H=1024, E=512, V=32000, L=3, RESCALE=4
// Strategy:
//   * All GEMMs use v_wmma_f32_16x16x32_f16 (f16 A/B, f32 accumulate).
//   * Per layer: hoist x@Wih^T + (bih+bhh) into one batched WMMA GEMM (gx).
//   * Recurrence: single 1024-thread workgroup, h (padded 8->16 rows) as f16
//     in LDS, c in registers; 32 waves x 32 H-cols; i/f/g/o tiles for the same
//     (b, col) live in matching lanes of 8 WMMA accumulators -> gate math is
//     pure register VALU, only barriers around the h swap.
//   * Final: tanh(cat@W1^T) -> f16, then @W2^T -> f32 logits (262 MB write).
// ---------------------------------------------------------------------------

typedef _Float16 half8 __attribute__((ext_vector_type(8)));
typedef _Float16 v16h  __attribute__((ext_vector_type(16)));
typedef float    v8f   __attribute__((ext_vector_type(8)));

#define WMMA_F16(a, b, c) \
  __builtin_amdgcn_wmma_f32_16x16x32_f16(false, (a), false, (b), (short)0, (c), false, false)

// A fragment (16x32, f16): lanes 0-15 -> row M=lane, K {k0..k0+7, k0+16..k0+23};
// lanes 16-31 -> row M=lane-16, K {k0+8..k0+15, k0+24..k0+31}.  (ISA 7.12.2)
static __device__ __forceinline__ v16h frag_a16(const _Float16* __restrict__ A,
                                                int lda, int k0, int lane) {
  int r  = lane & 15;
  int kc = k0 + ((lane & 16) >> 1);           // +8 for upper half-wave
  const _Float16* p = A + (size_t)r * lda + kc;
  half8 lo = *(const half8*)(p);
  half8 hi = *(const half8*)(p + 16);
  v16h v;
#pragma unroll
  for (int i = 0; i < 8; ++i) { v[i] = lo[i]; v[i + 8] = hi[i]; }
  return v;
}

// B fragment (32x16, f16), weights stored row-major [N,K] (i.e. W so y=x@W^T):
// lanes 0-15 -> col N=lane, K {k0..k0+7, k0+16..k0+23}; lanes 16-31 -> col N=lane-16,
// K {k0+8..k0+15, k0+24..k0+31}. Same K order as the A fragment.
static __device__ __forceinline__ v16h frag_b16(const _Float16* __restrict__ W,
                                                int ldw, int k0, int lane) {
  int r  = lane & 15;
  int kc = k0 + ((lane & 16) >> 1);           // +8 for upper half-wave
  const _Float16* p = W + (size_t)r * ldw + kc;
  half8 lo = *(const half8*)(p);
  half8 hi = *(const half8*)(p + 16);
  v16h v;
#pragma unroll
  for (int i = 0; i < 8; ++i) { v[i] = lo[i]; v[i + 8] = hi[i]; }
  return v;
}

// ---------------------------------------------------------------------------
// elementwise f32 -> f16
__global__ void dlm_f32_to_f16(const float* __restrict__ in,
                               _Float16* __restrict__ out, size_t n) {
  size_t i = blockIdx.x * (size_t)blockDim.x + threadIdx.x;
  size_t stride = gridDim.x * (size_t)blockDim.x;
  for (; i < n; i += stride) out[i] = (_Float16)in[i];
}

// embedding gather: x0[(s*8+b), e] = (f16) emb[src[b,s], e]
__global__ void dlm_embed(const int* __restrict__ src,
                          const float* __restrict__ emb,
                          _Float16* __restrict__ x0) {
  int bid = blockIdx.x;                 // = s*8 + b
  int s = bid >> 3, b = bid & 7;
  int tok = src[b * 256 + s];
  const float* e = emb + (size_t)tok * 512;
  _Float16* o = x0 + (size_t)bid * 512;
  for (int i = threadIdx.x; i < 512; i += blockDim.x) o[i] = (_Float16)e[i];
}

// ---------------------------------------------------------------------------
// Generic WMMA GEMM:  C[m,n] = epi( A[m,:] . W[n,:] + bias0[n] (+ bias1[n]) )
// A: f16 [M,lda], W: f16 [N,K] row-major.  Block = 128 threads = 4 waves;
// block tile = 16M x 256N, each wave owns 16x64 (4 accumulator tiles).
template <bool TANH_TO_F16>
__global__ __launch_bounds__(128) void dlm_gemm(
    const _Float16* __restrict__ A, int lda,
    const _Float16* __restrict__ W, int K,
    const float* __restrict__ bias0, const float* __restrict__ bias1,
    float* __restrict__ Cf, _Float16* __restrict__ Ch, int ldc) {
  const int lane = threadIdx.x & 31;
  const int wave = threadIdx.x >> 5;
  const int m0 = blockIdx.y << 4;
  const int n0 = (blockIdx.x << 8) + (wave << 6);

  const _Float16* Arow = A + (size_t)m0 * lda;
  const v8f vz = {0.f, 0.f, 0.f, 0.f, 0.f, 0.f, 0.f, 0.f};
  v8f acc0 = vz, acc1 = vz, acc2 = vz, acc3 = vz;

  for (int k0 = 0; k0 < K; k0 += 32) {
    v16h a  = frag_a16(Arow, lda, k0, lane);
    v16h b0 = frag_b16(W + (size_t)(n0 +  0) * K, K, k0, lane);
    v16h b1 = frag_b16(W + (size_t)(n0 + 16) * K, K, k0, lane);
    v16h b2 = frag_b16(W + (size_t)(n0 + 32) * K, K, k0, lane);
    v16h b3 = frag_b16(W + (size_t)(n0 + 48) * K, K, k0, lane);
    acc0 = WMMA_F16(a, b0, acc0);
    acc1 = WMMA_F16(a, b1, acc1);
    acc2 = WMMA_F16(a, b2, acc2);
    acc3 = WMMA_F16(a, b3, acc3);
  }

  const int rbase = m0 + ((lane & 16) >> 1);   // +8 for upper half-wave
  const int cin = lane & 15;
  v8f accs[4] = {acc0, acc1, acc2, acc3};
#pragma unroll
  for (int t = 0; t < 4; ++t) {
    int col = n0 + (t << 4) + cin;
    float bsum = bias0[col] + (bias1 ? bias1[col] : 0.0f);
#pragma unroll
    for (int r = 0; r < 8; ++r) {
      float v = accs[t][r] + bsum;
      size_t idx = (size_t)(rbase + r) * ldc + col;
      if (TANH_TO_F16) Ch[idx] = (_Float16)tanhf(v);
      else             Cf[idx] = v;
    }
  }
}

// ---------------------------------------------------------------------------
// LSTM recurrence for one layer, whole sequence, ONE workgroup (1024 thr).
// gx   : [S*B, 4H] f32, = x@Wih^T + bih + bhh, rows m = s*8+b
// whh  : [4H, H] f16
// h/c0 : [B, H] f32 (layer slice)
// xnext: [S*B, H] f16 (input to next layer, rows s*8+b)
// cat  : f16, already offset by layer*H; row stride 3H, rows b*S+s
__global__ __launch_bounds__(1024) void dlm_lstm_scan(
    const float* __restrict__ gx, const _Float16* __restrict__ whh,
    const float* __restrict__ fix, const float* __restrict__ h0,
    const float* __restrict__ c0, _Float16* __restrict__ xnext,
    _Float16* __restrict__ cat, int layer) {
  extern __shared__ _Float16 h16[];            // [16][1024] f16, rows 8..15 = 0 pad
  const int tid  = threadIdx.x;
  const int lane = tid & 31;
  const int wave = tid >> 5;
  const int colbase = wave << 5;               // each wave owns 32 H-columns

  for (int i = tid; i < 16 * 1024; i += 1024) {
    int r = i >> 10, c = i & 1023;
    h16[i] = (r < 8) ? (_Float16)h0[r * 1024 + c] : (_Float16)0.0f;
  }
  float creg[2][8];
  if (lane < 16) {
#pragma unroll
    for (int t = 0; t < 2; ++t)
#pragma unroll
      for (int r = 0; r < 8; ++r)
        creg[t][r] = c0[r * 1024 + colbase + (t << 4) + lane];
  }
  __syncthreads();

  const v8f vz = {0.f, 0.f, 0.f, 0.f, 0.f, 0.f, 0.f, 0.f};
  for (int s = 0; s < 256; ++s) {
    v8f acc[4][2];
#pragma unroll
    for (int g = 0; g < 4; ++g) { acc[g][0] = vz; acc[g][1] = vz; }

    for (int k0 = 0; k0 < 1024; k0 += 32) {
      v16h a = frag_a16(h16, 1024, k0, lane);  // shared across all 8 tiles
#pragma unroll
      for (int g = 0; g < 4; ++g) {
#pragma unroll
        for (int t = 0; t < 2; ++t) {
          int n = (g << 10) + colbase + (t << 4);
          v16h b = frag_b16(whh + (size_t)n * 1024, 1024, k0, lane);
          acc[g][t] = WMMA_F16(a, b, acc[g][t]);
        }
      }
    }
    __syncthreads();                           // all reads of h16 done

    if (lane < 16) {                           // rows 0..7 (real batch) only
#pragma unroll
      for (int t = 0; t < 2; ++t) {
        int col = colbase + (t << 4) + lane;
#pragma unroll
        for (int r = 0; r < 8; ++r) {
          const float* grow = gx + ((size_t)(s * 8 + r) << 12);
          float ii = acc[0][t][r] + grow[col];
          float ff = acc[1][t][r] + grow[1024 + col];
          float gg = acc[2][t][r] + grow[2048 + col];
          float oo = acc[3][t][r] + grow[3072 + col];
          float si = 1.0f / (1.0f + __expf(-ii));
          float sf = 1.0f / (1.0f + __expf(-ff));
          float so = 1.0f / (1.0f + __expf(-oo));
          float cn = sf * creg[t][r] + si * tanhf(gg);
          float hn = so * tanhf(cn);
          float fp = floorf(fix[r * 256 + s] * 0.25f);    // RESCALE = 4
          bool keep = ((float)layer > fp);                 // mask==1 -> keep old
          float hold = (float)h16[r * 1024 + col];
          float cb = keep ? creg[t][r] : cn;
          float hb = keep ? hold : hn;
          creg[t][r] = cb;
          _Float16 hbf = (_Float16)hb;
          h16[r * 1024 + col] = hbf;
          xnext[((size_t)(s * 8 + r) << 10) + col] = hbf;
          cat[(size_t)(r * 256 + s) * 3072 + col] = hbf;
        }
      }
    }
    __syncthreads();                           // h16 update visible to all
  }
}

// ---------------------------------------------------------------------------
extern "C" void kernel_launch(void* const* d_in, const int* in_sizes, int n_in,
                              void* d_out, int out_size, void* d_ws, size_t ws_size,
                              hipStream_t stream) {
  (void)in_sizes; (void)n_in; (void)out_size; (void)ws_size;
  const int*   src  = (const int*)  d_in[0];
  const float* fix  = (const float*)d_in[1];
  const float* h0   = (const float*)d_in[2];
  const float* c0   = (const float*)d_in[3];
  const float* emb  = (const float*)d_in[4];
  const float* Wih0 = (const float*)d_in[5];
  const float* Wihr = (const float*)d_in[6];
  const float* Whh  = (const float*)d_in[7];
  const float* bih  = (const float*)d_in[8];
  const float* bhh  = (const float*)d_in[9];
  const float* W1   = (const float*)d_in[10];
  const float* b1   = (const float*)d_in[11];
  const float* W2   = (const float*)d_in[12];
  const float* b2   = (const float*)d_in[13];
  float* out = (float*)d_out;

  char* base = (char*)d_ws;
  size_t off = 0;
  auto take = [&](size_t bytes) -> char* {
    char* p = base + off;
    off = (off + bytes + 255) & ~(size_t)255;
    return p;
  };
  // ~141 MB total workspace
  _Float16* w2h   = (_Float16*)take((size_t)32000 * 512 * 2);
  _Float16* w1h   = (_Float16*)take((size_t)512 * 3072 * 2);
  _Float16* wih0h = (_Float16*)take((size_t)4096 * 512 * 2);
  _Float16* wihrh = (_Float16*)take((size_t)2 * 4096 * 1024 * 2);
  _Float16* whhh  = (_Float16*)take((size_t)3 * 4096 * 1024 * 2);
  _Float16* x0h   = (_Float16*)take((size_t)2048 * 512 * 2);
  _Float16* xA    = (_Float16*)take((size_t)2048 * 1024 * 2);
  _Float16* xB    = (_Float16*)take((size_t)2048 * 1024 * 2);
  float*    gxf   = (float*)   take((size_t)2048 * 4096 * 4);
  _Float16* cath  = (_Float16*)take((size_t)2048 * 3072 * 2);
  _Float16* hidh  = (_Float16*)take((size_t)2048 * 512 * 2);

  // weight conversions (f32 -> f16), ~10 us of HBM traffic total
  dlm_f32_to_f16<<<2048, 256, 0, stream>>>(W2,   w2h,   (size_t)32000 * 512);
  dlm_f32_to_f16<<<2048, 256, 0, stream>>>(W1,   w1h,   (size_t)512 * 3072);
  dlm_f32_to_f16<<<2048, 256, 0, stream>>>(Wih0, wih0h, (size_t)4096 * 512);
  dlm_f32_to_f16<<<2048, 256, 0, stream>>>(Wihr, wihrh, (size_t)2 * 4096 * 1024);
  dlm_f32_to_f16<<<2048, 256, 0, stream>>>(Whh,  whhh,  (size_t)3 * 4096 * 1024);

  dlm_embed<<<2048, 256, 0, stream>>>(src, emb, x0h);

  const _Float16* xin = x0h;
  int Kin = 512;
  _Float16* ping[2] = {xA, xB};
  for (int l = 0; l < 3; ++l) {
    const _Float16* wih = (l == 0) ? wih0h : (wihrh + (size_t)(l - 1) * 4096 * 1024);
    // gx = x @ Wih^T + bih + bhh   ([2048,Kin] x [Kin,4096])
    dim3 g1(4096 / 256, 2048 / 16);
    dlm_gemm<false><<<g1, 128, 0, stream>>>(xin, Kin, wih, Kin,
                                            bih + l * 4096, bhh + l * 4096,
                                            gxf, nullptr, 4096);
    // sequential recurrence, one workgroup, 32 KB dynamic LDS for h
    _Float16* xo = ping[l & 1];
    dlm_lstm_scan<<<1, 1024, 16 * 1024 * 2, stream>>>(
        gxf, whhh + (size_t)l * 4096 * 1024, fix,
        h0 + (size_t)l * 8 * 1024, c0 + (size_t)l * 8 * 1024,
        xo, cath + l * 1024, l);
    xin = xo;
    Kin = 1024;
  }

  // hidden = tanh(cat @ W1^T + b1) -> f16   ([2048,3072] x [3072,512])
  {
    dim3 g2(512 / 256, 2048 / 16);
    dlm_gemm<true><<<g2, 128, 0, stream>>>(cath, 3072, w1h, 3072,
                                           b1, nullptr, nullptr, hidh, 512);
  }
  // prediction = hidden @ W2^T + b2 -> f32  ([2048,512] x [512,32000])
  {
    dim3 g3(32000 / 256, 2048 / 16);
    dlm_gemm<false><<<g3, 128, 0, stream>>>(hidh, 512, w2h, 512,
                                            b2, nullptr, out, nullptr, 32000);
  }
}
